// mLSTMBackend_32392643346564
// MI455X (gfx1250) — matrix operand order repaired and checked by reading, not compiled, not run on hardware
//
#include <hip/hip_runtime.h>

typedef __attribute__((ext_vector_type(16))) __bf16 v16bf;
typedef __attribute__((ext_vector_type(8)))  float  v8f;

namespace {

constexpr int kB = 4, kNH = 8, kS = 4096, kDH = 128;
constexpr int kCL = 64;               // chunk length
constexpr int kNCH = kS / kCL;        // 64 chunks per head
constexpr int kThreads = 512;         // 16 wave32 waves
constexpr float kEps = 1e-6f;

struct SMem {
  // raw f32 staging for async global->LDS copies (next/current chunk)
  float Qf[kCL * kDH];             // 32 KB
  float Kf[kCL * kDH];             // 32 KB
  float Vf[kCL * kDH];             // 32 KB
  unsigned short Qb[kCL * kDH];    // [j][d] bf16
  unsigned short Kb[kCL * kDH];    // [s][d] bf16
  unsigned short Vt[kDH * kCL];    // [v][s] bf16 (V transposed)
  unsigned short Kgt[kDH * kCL];   // [k][s] = g_s * K[s][k] bf16
  unsigned short Cb[kDH * kDH];    // [v][k] bf16 copy of C state
  unsigned short Pb[kCL * kCL];    // [j][s] = (Q K^T) * D, causal, bf16
  float a[kCL], mr[kCL], iv[kCL], fv[kCL], dj[kCL], g[kCL];
  float denNq[kCL], denf[kCL];
  float Nv[kDH];
  float dL;
};

union BFV { unsigned int u[8]; v16bf v; };
union FV  { float f[8]; v8f v; };

__device__ __forceinline__ unsigned short f2bf(float x) {
  unsigned u = __float_as_uint(x);
  u += 0x7fffu + ((u >> 16) & 1u);   // round-to-nearest-even
  return (unsigned short)(u >> 16);
}
__device__ __forceinline__ float bf2f(unsigned short s) {
  return __uint_as_float((unsigned)s << 16);
}

// Low 32 bits of a generic pointer into LDS == DS-relative byte address
// (flat aperture rule).
__device__ __forceinline__ unsigned lds_off(const void* p) {
  return (unsigned)(uintptr_t)p;
}

// GLOBAL_LOAD_ASYNC_TO_LDS_B128: per-lane 16B copy global -> LDS, ASYNCcnt.
__device__ __forceinline__ void async_load_b128(unsigned dst_lds,
                                                const void* src) {
  asm volatile("global_load_async_to_lds_b128 %0, %1, off"
               :: "v"(dst_lds), "v"(src)
               : "memory");
}
__device__ __forceinline__ void wait_async0() {
  asm volatile("s_wait_asynccnt 0x0" ::: "memory");
}

// A operand (16x32 bf16). Lane l<16: M=l, K in {0..7} u {16..23};
// lane>=16: same M, K shifted by +8. VGPR r packs two consecutive K.
__device__ __forceinline__ v16bf load_a(const unsigned short* p, int stride,
                                        int m0, int k0, int lane) {
  const int half = (lane >> 4) & 1;
  const unsigned short* r = p + (m0 + (lane & 15)) * stride + k0 + half * 8;
  BFV t;
#pragma unroll
  for (int i = 0; i < 4; ++i) t.u[i] = *(const unsigned int*)(r + 2 * i);
#pragma unroll
  for (int i = 0; i < 4; ++i) t.u[4 + i] = *(const unsigned int*)(r + 16 + 2 * i);
  return t.v;
}
// B operand (32x16 bf16) read from B^T stored row-major [n][k].
// Lane l<16: N=l, K in {0..7} u {16..23}; lane>=16: N=l-16, K shifted by +8.
__device__ __forceinline__ v16bf load_b(const unsigned short* p, int stride,
                                        int n0, int k0, int lane) {
  const int half = (lane >> 4) & 1;
  const unsigned short* r = p + (n0 + (lane & 15)) * stride + k0 + half * 8;
  BFV t;
#pragma unroll
  for (int i = 0; i < 4; ++i) t.u[i] = *(const unsigned int*)(r + 2 * i);
#pragma unroll
  for (int i = 0; i < 4; ++i) t.u[4 + i] = *(const unsigned int*)(r + 16 + 2 * i);
  return t.v;
}

// Issue the async staging loads for one chunk (each thread loads the same
// slice it will later convert, so s_wait_asynccnt 0 alone makes it visible).
__device__ __forceinline__ void issue_chunk_loads(SMem& sm,
                                                  const float* __restrict__ qp,
                                                  const float* __restrict__ kp,
                                                  const float* __restrict__ vp,
                                                  int t0, int tid) {
  const float* qs = qp + (size_t)t0 * kDH;
  const float* ks = kp + (size_t)t0 * kDH;
  const float* vs = vp + (size_t)t0 * kDH;
#pragma unroll
  for (int e = 0; e < 4; ++e) {
    const int idx = (tid + e * kThreads) * 4;  // float index, 16B granules
    async_load_b128(lds_off(&sm.Qf[idx]), qs + idx);
    async_load_b128(lds_off(&sm.Kf[idx]), ks + idx);
    async_load_b128(lds_off(&sm.Vf[idx]), vs + idx);
  }
}

__global__ __launch_bounds__(kThreads, 1) void mlstm_chunkwise_wmma(
    const float* __restrict__ q, const float* __restrict__ k,
    const float* __restrict__ v, const float* __restrict__ ig,
    const float* __restrict__ fg, float* __restrict__ out) {
  extern __shared__ __align__(16) unsigned char smraw[];
  SMem& sm = *reinterpret_cast<SMem*>(smraw);

  const int tid = threadIdx.x;
  const int lane = tid & 31;
  const int w = tid >> 5;          // wave id, 0..15 (wave32)
  const int bh = blockIdx.x;

  const float* qp = q + (size_t)bh * kS * kDH;
  const float* kp = k + (size_t)bh * kS * kDH;
  const float* vp = v + (size_t)bh * kS * kDH;
  const float* ip = ig + (size_t)bh * kS;
  const float* fp = fg + (size_t)bh * kS;
  float* op = out + (size_t)bh * kS * kDH;

  // ---- kick off async staging of chunk 0 ----
  issue_chunk_loads(sm, qp, kp, vp, 0, tid);

  // ---- initial state: C=0 (registers + bf16 LDS copy), N=1, m=0 ----
  v8f cacc[4];
#pragma unroll
  for (int c = 0; c < 4; ++c) {
    FV z;
#pragma unroll
    for (int r = 0; r < 8; ++r) z.f[r] = 0.f;
    cacc[c] = z.v;
  }
  for (int idx = tid; idx < kDH * kDH / 2; idx += kThreads)
    ((unsigned int*)sm.Cb)[idx] = 0u;
  if (tid < kDH) sm.Nv[tid] = 1.0f;
  float m_carry = 0.0f;  // meaningful on thread 0 only
  __syncthreads();

#pragma clang loop unroll(disable)
  for (int ch = 0; ch < kNCH; ++ch) {
    const int t0 = ch * kCL;

    // ---- gate scan (cheap, scalar; overlaps in-flight async loads) ----
    if (tid < kCL) { sm.iv[tid] = ip[t0 + tid]; sm.fv[tid] = fp[t0 + tid]; }
    __syncthreads();
    if (tid == 0) {
      float aa = 0.f, mprev = m_carry;
#pragma clang loop unroll(disable)
      for (int j = 0; j < kCL; ++j) {
        aa += sm.fv[j];
        sm.a[j] = aa;
        const float mm = fmaxf(mprev + sm.fv[j], sm.iv[j]);
        sm.mr[j] = mm;
        mprev = mm;
        sm.dj[j] = __expf(m_carry + aa - mm);
      }
      const float aL = aa, mL = mprev;
      sm.dL = __expf(m_carry + aL - mL);
#pragma clang loop unroll(disable)
      for (int s = 0; s < kCL; ++s)
        sm.g[s] = __expf(aL - sm.a[s] + sm.iv[s] - mL);
      m_carry = mL;
    }
    __syncthreads();

    // ---- wait for this chunk's staged data (own slice only needed) ----
    wait_async0();

    // ---- convert staged f32 -> bf16 LDS tiles (K,V also transposed) ----
    {
      const float4* q4 = (const float4*)sm.Qf;
      const float4* k4 = (const float4*)sm.Kf;
      const float4* v4 = (const float4*)sm.Vf;
#pragma unroll
      for (int e = 0; e < 4; ++e) {
        const int idx = tid + e * kThreads;    // 2048 float4 per matrix
        const int srow = idx >> 5;             // time index within chunk
        const int dcol = (idx & 31) * 4;       // feature index
        float4 x = q4[idx];
        *(unsigned int*)&sm.Qb[srow * kDH + dcol] =
            (unsigned)f2bf(x.x) | ((unsigned)f2bf(x.y) << 16);
        *(unsigned int*)&sm.Qb[srow * kDH + dcol + 2] =
            (unsigned)f2bf(x.z) | ((unsigned)f2bf(x.w) << 16);
        float4 y = k4[idx];
        *(unsigned int*)&sm.Kb[srow * kDH + dcol] =
            (unsigned)f2bf(y.x) | ((unsigned)f2bf(y.y) << 16);
        *(unsigned int*)&sm.Kb[srow * kDH + dcol + 2] =
            (unsigned)f2bf(y.z) | ((unsigned)f2bf(y.w) << 16);
        const float gs = sm.g[srow];
        sm.Kgt[(dcol + 0) * kCL + srow] = f2bf(gs * y.x);
        sm.Kgt[(dcol + 1) * kCL + srow] = f2bf(gs * y.y);
        sm.Kgt[(dcol + 2) * kCL + srow] = f2bf(gs * y.z);
        sm.Kgt[(dcol + 3) * kCL + srow] = f2bf(gs * y.w);
        float4 z = v4[idx];
        sm.Vt[(dcol + 0) * kCL + srow] = f2bf(z.x);
        sm.Vt[(dcol + 1) * kCL + srow] = f2bf(z.y);
        sm.Vt[(dcol + 2) * kCL + srow] = f2bf(z.z);
        sm.Vt[(dcol + 3) * kCL + srow] = f2bf(z.w);
      }
    }
    __syncthreads();  // drains DScnt; staging is now free to refill

    // ---- prefetch next chunk behind the compute phases ----
    if (ch + 1 < kNCH) issue_chunk_loads(sm, qp, kp, vp, t0 + kCL, tid);

    // ---- inter denominator term: N_old . q_j ----
    if (tid < kCL) {
      float acc = 0.f;
      for (int d = 0; d < kDH; ++d)
        acc += sm.Nv[d] * bf2f(sm.Qb[tid * kDH + d]);
      sm.denNq[tid] = acc;
    }

    // ---- S = Q K^T (one 16x16 tile per wave), apply causal decay -> P ----
    {
      const int jt = w >> 2, st = w & 3;
      FV acc;
#pragma unroll
      for (int r = 0; r < 8; ++r) acc.f[r] = 0.f;
#pragma unroll
      for (int kk = 0; kk < 4; ++kk) {
        v16bf A = load_a(sm.Qb, kDH, jt * 16, kk * 32, lane);
        v16bf Bm = load_b(sm.Kb, kDH, st * 16, kk * 32, lane);
        acc.v = __builtin_amdgcn_wmma_f32_16x16x32_bf16(
            false, A, false, Bm, (short)0, acc.v, false, false);
      }
      const int nn = lane & 15;
      const int mb = (lane < 16) ? 0 : 8;
#pragma unroll
      for (int r = 0; r < 8; ++r) {
        const int j = jt * 16 + mb + r;
        const int s = st * 16 + nn;
        float pv = 0.f;
        if (s <= j)
          pv = acc.f[r] * __expf(sm.a[j] - sm.a[s] + sm.iv[s] - sm.mr[j]);
        sm.Pb[j * kCL + s] = f2bf(pv);
      }
    }
    __syncthreads();

    // ---- final denominator: max(dj*(N.q) + rowsum(P), EPS) ----
    if (tid < kCL) {
      float rs = 0.f;
      for (int s = 0; s < kCL; ++s) rs += bf2f(sm.Pb[tid * kCL + s]);
      sm.denf[tid] = fmaxf(sm.dj[tid] * sm.denNq[tid] + rs, kEps);
    }
    __syncthreads();

    // ---- H = dj*(Q C^T) + P V, divide by denom; 2 tiles per wave ----
#pragma unroll
    for (int hi = 0; hi < 2; ++hi) {
      const int id = w * 2 + hi;
      const int jt = id >> 3, vt = id & 7;
      FV acc;
#pragma unroll
      for (int r = 0; r < 8; ++r) acc.f[r] = 0.f;
#pragma unroll
      for (int kk = 0; kk < 4; ++kk) {
        v16bf A = load_a(sm.Qb, kDH, jt * 16, kk * 32, lane);
        v16bf Bm = load_b(sm.Cb, kDH, vt * 16, kk * 32, lane);
        acc.v = __builtin_amdgcn_wmma_f32_16x16x32_bf16(
            false, A, false, Bm, (short)0, acc.v, false, false);
      }
      const int nn = lane & 15;
      const int mb = (lane < 16) ? 0 : 8;
#pragma unroll
      for (int r = 0; r < 8; ++r) acc.f[r] *= sm.dj[jt * 16 + mb + r];
#pragma unroll
      for (int ss = 0; ss < 2; ++ss) {
        v16bf A = load_a(sm.Pb, kCL, jt * 16, ss * 32, lane);
        v16bf Bm = load_b(sm.Vt, kCL, vt * 16, ss * 32, lane);
        acc.v = __builtin_amdgcn_wmma_f32_16x16x32_bf16(
            false, A, false, Bm, (short)0, acc.v, false, false);
      }
#pragma unroll
      for (int r = 0; r < 8; ++r) {
        const int j = jt * 16 + mb + r;
        op[(size_t)(t0 + j) * kDH + vt * 16 + nn] = acc.f[r] / sm.denf[j];
      }
    }
    __syncthreads();

    // ---- state update: C = dL*C + V^T diag(g) K (register resident) ----
    const float dL = sm.dL;
#pragma unroll
    for (int ci = 0; ci < 4; ++ci) {
      const int id = w * 4 + ci;
      const int vt = id >> 3, kt = id & 7;
      FV acc;
      acc.v = cacc[ci];
#pragma unroll
      for (int r = 0; r < 8; ++r) acc.f[r] *= dL;
#pragma unroll
      for (int ss = 0; ss < 2; ++ss) {
        v16bf A = load_a(sm.Vt, kCL, vt * 16, ss * 32, lane);
        v16bf Bm = load_b(sm.Kgt, kCL, kt * 16, ss * 32, lane);
        acc.v = __builtin_amdgcn_wmma_f32_16x16x32_bf16(
            false, A, false, Bm, (short)0, acc.v, false, false);
      }
      cacc[ci] = acc.v;
      const int nn = lane & 15;
      const int mb = (lane < 16) ? 0 : 8;
#pragma unroll
      for (int r = 0; r < 8; ++r)
        sm.Cb[(vt * 16 + mb + r) * kDH + kt * 16 + nn] = f2bf(acc.f[r]);
    }
    // ---- N update: N = dL*N + sum_s g_s K[s][:] (Kgt already scaled) ----
    if (tid < kDH) {
      float acc = 0.f;
      for (int s = 0; s < kCL; ++s) acc += bf2f(sm.Kgt[tid * kCL + s]);
      sm.Nv[tid] = dL * sm.Nv[tid] + acc;
    }
    __syncthreads();
  }
}

}  // namespace

extern "C" void kernel_launch(void* const* d_in, const int* in_sizes, int n_in,
                              void* d_out, int out_size, void* d_ws,
                              size_t ws_size, hipStream_t stream) {
  (void)in_sizes; (void)n_in; (void)out_size; (void)d_ws; (void)ws_size;
  const float* q = (const float*)d_in[0];
  const float* k = (const float*)d_in[1];
  const float* v = (const float*)d_in[2];
  const float* ig = (const float*)d_in[3];
  const float* fg = (const float*)d_in[4];
  float* out = (float*)d_out;
  dim3 grid(kB * kNH), block(kThreads);
  mlstm_chunkwise_wmma<<<grid, block, sizeof(SMem), stream>>>(q, k, v, ig, fg,
                                                              out);
}
